// CustomTransformerLayer_54494545052072
// MI455X (gfx1250) — compile-verified, matrix operand order checked by reading, not run on hardware
//
#include <hip/hip_runtime.h>

// ---------------------------------------------------------------------------
// Types for CDNA5 WMMA / TDM (wave32)
// ---------------------------------------------------------------------------
typedef __attribute__((ext_vector_type(16))) __bf16       v16bf;
typedef __attribute__((ext_vector_type(4)))  __bf16       v4bf;
typedef __attribute__((ext_vector_type(8)))  float        v8f;
typedef __attribute__((ext_vector_type(4)))  float        v4f;
typedef __attribute__((ext_vector_type(4)))  int          v4i;
typedef __attribute__((ext_vector_type(8)))  int          v8i;
typedef __attribute__((ext_vector_type(4)))  unsigned int v4u;

union FragAB { v16bf v; uint4 u[2]; };
union FragC  { v8f  v; float f[8]; };

// ---------------------------------------------------------------------------
// Model constants
// ---------------------------------------------------------------------------
#define SEQ     2048
#define HID     4096
#define FFN_DIM 16384
#define NHEAD   32
#define D_NOPE  128
#define D_ROPE  64
#define D_V     128
#define QK_DIM  (NHEAD * (D_NOPE + D_ROPE))   // 6144
#define HEAD_K  (D_NOPE + D_ROPE)             // 192

// ---------------------------------------------------------------------------
// bf16 conversion via vector converts (emits v_cvt_pk_bf16_f32 cleanly)
// ---------------------------------------------------------------------------
__device__ __forceinline__ unsigned short f2bf(float a) {
    union { __bf16 h; unsigned short u; } x;
    x.h = (__bf16)a;
    return x.u;
}
__device__ __forceinline__ uint2 pk4_bf16(float4 f) {
    union { v4bf h; uint2 u; } cv;
    v4f vf = {f.x, f.y, f.z, f.w};
    cv.h = __builtin_convertvector(vf, v4bf);
    return cv.u;
}

// ---------------------------------------------------------------------------
// Async global -> LDS (ASYNCcnt path), with sync fallback
// ---------------------------------------------------------------------------
#if defined(__AMDGCN__) && __has_builtin(__builtin_amdgcn_global_load_async_to_lds_b128)
#define USE_ASYNC 1
#else
#define USE_ASYNC 0
#endif

// Tensor Data Mover (TENSORcnt path)
#if defined(__AMDGCN__) && __has_builtin(__builtin_amdgcn_tensor_load_to_lds)
#define USE_TDM 1
#else
#define USE_TDM 0
#endif

__device__ __forceinline__ void g2l_16B(const void* g, void* l) {
#if USE_ASYNC
    __builtin_amdgcn_global_load_async_to_lds_b128(
        (__attribute__((address_space(1))) v4i*)(g),
        (__attribute__((address_space(3))) v4i*)(l), 0, 0);
#else
    *(uint4*)l = *(const uint4*)g;
#endif
}

__device__ __forceinline__ void wait_async0() {
#if USE_ASYNC
#if __has_builtin(__builtin_amdgcn_s_wait_asynccnt)
    __builtin_amdgcn_s_wait_asynccnt(0);
#else
    asm volatile("s_wait_asynccnt 0x0" ::: "memory");
#endif
#endif
}

__device__ __forceinline__ void wait_tensor0() {
#if USE_TDM
#if __has_builtin(__builtin_amdgcn_s_wait_tensorcnt)
    __builtin_amdgcn_s_wait_tensorcnt(0);
#else
    asm volatile("s_wait_tensorcnt 0x0" ::: "memory");
#endif
#endif
}

#if USE_TDM
// One-shot 2D TDM load: rows x cols bf16 tile, row stride = strideElems.
// D# per CDNA5 ISA 8.3/8.4: group0 = {count/type/addr}, group1 = {dims/strides}.
// This toolchain's builtin is the 6-arg form: (g0, g1, g2, g3, g4, cpol).
__device__ __forceinline__ void tdm_load_tile_bf16(const void* gsrc, unsigned ldsByteAddr,
                                                   unsigned rows, unsigned cols,
                                                   unsigned strideElems)
{
    unsigned long long ga = (unsigned long long)gsrc;
    v4u g0;
    g0.x = 1u;                                                 // count=1 (valid user D#)
    g0.y = ldsByteAddr;                                        // lds_addr
    g0.z = (unsigned)(ga & 0xFFFFFFFFu);                       // global_addr[31:0]
    g0.w = (unsigned)((ga >> 32) & 0x01FFFFFFu) | (2u << 30);  // global_addr[56:32] | type=2
    v8i g1;
    g1[0] = (int)(1u << 16);                                   // data_size=1 (2 bytes)
    g1[1] = (int)((cols & 0xFFFFu) << 16);                     // tensor_dim0[15:0]
    g1[2] = (int)(((cols >> 16) & 0xFFFFu) | ((rows & 0xFFFFu) << 16)); // dim0 hi | tensor_dim1 lo
    g1[3] = (int)(((rows >> 16) & 0xFFFFu) | ((cols & 0xFFFFu) << 16)); // dim1 hi | tile_dim0
    g1[4] = (int)(rows & 0xFFFFu);                             // tile_dim1 (tile_dim2 = 0)
    g1[5] = (int)strideElems;                                  // tensor_dim0_stride[31:0]
    g1[6] = 0;
    g1[7] = 0;
    v4i z4 = {0, 0, 0, 0};
    v8i z8 = {0, 0, 0, 0, 0, 0, 0, 0};
    __builtin_amdgcn_tensor_load_to_lds(g0, g1, z4, z4, z8, 0);
}
__device__ __forceinline__ unsigned lds_addr_of(void* p) {
    return (unsigned)(size_t)(__attribute__((address_space(3))) char*)(p);
}
#endif

// ---------------------------------------------------------------------------
// WMMA GEMM:  C[M,N] = A[M,K] @ B[N,K]^T
//   A: bf16 row-major, staged via TDM (TENSORcnt) or per-lane async (ASYNCcnt)
//   B: fp32 (converted while staging) or bf16 (TDM/async)
//   EPI: 0 = store, 1 = store + residual add, 2 = SiLU
//   C_BF16: store C as bf16 (feeds the next GEMM's A side)
// Block: 256 threads (8 waves), tile 128x128, BK=32.
// ---------------------------------------------------------------------------
template<bool B_F32, int EPI, bool C_BF16>
__global__ void __launch_bounds__(256)
gemm_wmma_kernel(const unsigned short* __restrict__ A, int lda,
                 const void* __restrict__ Braw, int ldb,
                 void* __restrict__ Cv, int ldc,
                 const float* __restrict__ Rsd,
                 int M, int N, int K)
{
    __shared__ uint4 sA[512];   // 128 rows x 32 bf16 = 8KB
    __shared__ uint4 sB[512];   // 128 rows x 32 bf16 = 8KB

    const int tid      = threadIdx.x;
    const int lane     = tid & 31;
    const int wid      = tid >> 5;
    const int waveM    = wid & 3;        // 0..3
    const int waveN    = wid >> 2;       // 0..1
    const int loadRow  = tid >> 1;       // 0..127
    const int loadHalf = tid & 1;        // 0..1  (16 bf16 elements each)

    const int blockM = blockIdx.y * 128;
    const int blockN = blockIdx.x * 128;

    const v8f zero = {0.f,0.f,0.f,0.f,0.f,0.f,0.f,0.f};
    v8f acc[2][4];
#pragma unroll
    for (int m = 0; m < 2; ++m)
#pragma unroll
        for (int n = 0; n < 4; ++n) acc[m][n] = zero;

    const int aRow = blockM + loadRow;
    const int bRow = blockN + loadRow;
    const int sIdx = loadRow * 4 + loadHalf * 2;

    for (int k0 = 0; k0 < K; k0 += 32) {
        __syncthreads();                       // previous tile fully consumed
#if USE_TDM
        // ---- stage via Tensor Data Mover: one DMA per tile, issued by wave 0 ----
        if (wid == 0) {
            tdm_load_tile_bf16(A + (size_t)blockM * lda + k0,
                               lds_addr_of(&sA[0]), 128, 32, (unsigned)lda);
            if (!B_F32)
                tdm_load_tile_bf16((const unsigned short*)Braw + (size_t)blockN * ldb + k0,
                                   lds_addr_of(&sB[0]), 128, 32, (unsigned)ldb);
        }
#else
        {   // per-lane async global->LDS
            const unsigned short* agp = A + (size_t)aRow * lda + k0 + loadHalf * 16;
            g2l_16B(agp,     &sA[sIdx + 0]);
            g2l_16B(agp + 8, &sA[sIdx + 1]);
        }
        if (!B_F32) {
            const unsigned short* bgp =
                (const unsigned short*)Braw + (size_t)bRow * ldb + k0 + loadHalf * 16;
            g2l_16B(bgp,     &sB[sIdx + 0]);
            g2l_16B(bgp + 8, &sB[sIdx + 1]);
        }
#endif
        // ---- fp32 weight tile: load + convert to bf16 in VGPRs ----
        if (B_F32) {
            const float4* bp = reinterpret_cast<const float4*>(
                (const float*)Braw + (size_t)bRow * ldb + k0 + loadHalf * 16);
            float4 f0 = bp[0], f1 = bp[1], f2 = bp[2], f3 = bp[3];
            uint2 c0 = pk4_bf16(f0), c1 = pk4_bf16(f1), c2 = pk4_bf16(f2), c3 = pk4_bf16(f3);
            uint4 p0, p1;
            p0.x = c0.x; p0.y = c0.y; p0.z = c1.x; p0.w = c1.y;
            p1.x = c2.x; p1.y = c2.y; p1.z = c3.x; p1.w = c3.y;
            sB[sIdx + 0] = p0; sB[sIdx + 1] = p1;
            if (k0 + 32 < K)   // global_prefetch_b8 on next weight tile
                __builtin_prefetch((const float*)Braw + (size_t)bRow * ldb + k0 + 32 + loadHalf * 16, 0, 1);
        }
#if USE_TDM
        wait_tensor0();
#else
        wait_async0();
#endif
        __syncthreads();                       // tile visible to all waves

        // ---- fragments: 16-bit A layout (lanes 0-15: K 0-7/16-23, lanes 16-31: K 8-15/24-31) ----
        const int kh = lane >> 4;
        FragAB a[2], b[4];
#pragma unroll
        for (int m = 0; m < 2; ++m) {
            int r = waveM * 32 + m * 16 + (lane & 15);
            a[m].u[0] = sA[r * 4 + kh];
            a[m].u[1] = sA[r * 4 + 2 + kh];
        }
#pragma unroll
        for (int n = 0; n < 4; ++n) {
            int c = waveN * 64 + n * 16 + (lane & 15);
            b[n].u[0] = sB[c * 4 + kh];
            b[n].u[1] = sB[c * 4 + 2 + kh];
        }
#pragma unroll
        for (int m = 0; m < 2; ++m)
#pragma unroll
            for (int n = 0; n < 4; ++n)
                acc[m][n] = __builtin_amdgcn_wmma_f32_16x16x32_bf16(
                    false, a[m].v, false, b[n].v, (short)0, acc[m][n], false, false);
    }

    // ---- epilogue: f32 16x16 C/D layout (VGPR r: lanes<16 -> M=r, lanes>=16 -> M=8+r) ----
    const int lcol  = lane & 15;
    const int lrow8 = (lane >> 4) * 8;
#pragma unroll
    for (int m = 0; m < 2; ++m) {
#pragma unroll
        for (int n = 0; n < 4; ++n) {
            FragC fc; fc.v = acc[m][n];
            const int col = blockN + waveN * 64 + n * 16 + lcol;
#pragma unroll
            for (int r = 0; r < 8; ++r) {
                const int row = blockM + waveM * 32 + m * 16 + lrow8 + r;
                float val = fc.f[r];
                if (EPI == 2) val = val * (1.0f / (1.0f + __expf(-val)));   // SiLU
                if (EPI == 1) val += Rsd[(size_t)row * ldc + col];          // residual
                if (C_BF16)
                    ((unsigned short*)Cv)[(size_t)row * ldc + col] = f2bf(val);
                else
                    ((float*)Cv)[(size_t)row * ldc + col] = val;
            }
        }
    }
    (void)M; (void)N;
}

// ---------------------------------------------------------------------------
// LayerNorm: one block of 256 threads per row of 4096; bf16 output
// ---------------------------------------------------------------------------
__global__ void __launch_bounds__(256)
ln_kernel(const float* __restrict__ x, const float* __restrict__ w,
          const float* __restrict__ b, unsigned short* __restrict__ out)
{
    const int row = blockIdx.x, t = threadIdx.x;
    const float* xr = x + (size_t)row * HID;
    float vals[16];
    float s = 0.f, s2 = 0.f;
#pragma unroll
    for (int j = 0; j < 16; ++j) {
        float v = xr[t + j * 256];
        vals[j] = v; s += v; s2 += v * v;
    }
    __shared__ float r1[256], r2[256];
    r1[t] = s; r2[t] = s2; __syncthreads();
    for (int off = 128; off > 0; off >>= 1) {
        if (t < off) { r1[t] += r1[t + off]; r2[t] += r2[t + off]; }
        __syncthreads();
    }
    const float mu  = r1[0] * (1.0f / HID);
    const float var = r2[0] * (1.0f / HID) - mu * mu;
    const float inv = rsqrtf(var + 1e-6f);
#pragma unroll
    for (int j = 0; j < 16; ++j) {
        int c = t + j * 256;
        out[(size_t)row * HID + c] = f2bf((vals[j] - mu) * inv * w[c] + b[c]);
    }
}

// ---------------------------------------------------------------------------
// RoPE in place on q and k rope slices (D_ROPE=64, half=32)
// ---------------------------------------------------------------------------
__global__ void __launch_bounds__(256)
rope_kernel(float* __restrict__ q, float* __restrict__ k)
{
    const int idx = blockIdx.x * 256 + threadIdx.x;
    const int i = idx & 31;
    const int h = (idx >> 5) & 31;
    const int s = idx >> 10;
    if (s >= SEQ) return;
    const float inv_freq = __expf(-(float)i * (9.210340371976184f / 32.0f)); // 10000^(-2i/64)
    const float f  = (float)s * inv_freq;
    const float cs = cosf(f), sn = sinf(f);
    const size_t base = (size_t)s * QK_DIM + NHEAD * D_NOPE + (size_t)h * D_ROPE;
    float a0 = q[base + i], a1 = q[base + 32 + i];
    q[base + i]      = a0 * cs - a1 * sn;
    q[base + 32 + i] = a0 * sn + a1 * cs;
    float b0 = k[base + i], b1 = k[base + 32 + i];
    k[base + i]      = b0 * cs - b1 * sn;
    k[base + 32 + i] = b0 * sn + b1 * cs;
}

// ---------------------------------------------------------------------------
// Per-head gather: qh (scaled, 2048x192 bf16), kh (2048x192 bf16),
//                  vT (128x2048 bf16, transposed so ctx GEMM is N-major x K)
// ---------------------------------------------------------------------------
__global__ void __launch_bounds__(256)
gather_head_kernel(const float* __restrict__ q, const float* __restrict__ k,
                   const float* __restrict__ v,
                   unsigned short* __restrict__ qh, unsigned short* __restrict__ kh,
                   unsigned short* __restrict__ vt, int hd)
{
    const int idx = blockIdx.x * 256 + threadIdx.x;
    const int NQ = SEQ * HEAD_K;                 // 393216
    if (idx < NQ) {
        const int s = idx / HEAD_K, d = idx % HEAD_K;
        float val = (d < D_NOPE)
            ? q[(size_t)s * QK_DIM + hd * D_NOPE + d] * 0.0883883476483184f   // 1/sqrt(128)
            : q[(size_t)s * QK_DIM + NHEAD * D_NOPE + hd * D_ROPE + (d - D_NOPE)] * 0.125f; // 1/sqrt(64)
        qh[idx] = f2bf(val);
    } else if (idx < 2 * NQ) {
        const int j = idx - NQ;
        const int s = j / HEAD_K, d = j % HEAD_K;
        float val = (d < D_NOPE)
            ? k[(size_t)s * QK_DIM + hd * D_NOPE + d]
            : k[(size_t)s * QK_DIM + NHEAD * D_NOPE + hd * D_ROPE + (d - D_NOPE)];
        kh[j] = f2bf(val);
    } else {
        const int j = idx - 2 * NQ;
        if (j < D_V * SEQ) {
            const int d = j >> 11;          // /2048
            const int s = j & (SEQ - 1);
            vt[j] = f2bf(v[(size_t)s * (NHEAD * D_V) + hd * D_V + d]);
        }
    }
}

// ---------------------------------------------------------------------------
// Causal softmax: one block per row; outputs bf16 probabilities
// ---------------------------------------------------------------------------
__global__ void __launch_bounds__(256)
softmax_kernel(const float* __restrict__ S, unsigned short* __restrict__ P)
{
    const int row = blockIdx.x, t = threadIdx.x;
    const float* sr = S + (size_t)row * SEQ;
    unsigned short* pr = P + (size_t)row * SEQ;
    float vals[8];
    float m = -3.4e38f;
#pragma unroll
    for (int j = 0; j < 8; ++j) {
        int c = t + j * 256;
        float v = (c <= row) ? sr[c] : -3.4e38f;
        vals[j] = v; m = fmaxf(m, v);
    }
    __shared__ float red[256];
    red[t] = m; __syncthreads();
    for (int off = 128; off > 0; off >>= 1) {
        if (t < off) red[t] = fmaxf(red[t], red[t + off]);
        __syncthreads();
    }
    m = red[0]; __syncthreads();
    float s = 0.f;
#pragma unroll
    for (int j = 0; j < 8; ++j) {
        int c = t + j * 256;
        float e = (c <= row) ? __expf(vals[j] - m) : 0.f;
        vals[j] = e; s += e;
    }
    red[t] = s; __syncthreads();
    for (int off = 128; off > 0; off >>= 1) {
        if (t < off) red[t] += red[t + off];
        __syncthreads();
    }
    const float inv = 1.0f / red[0];
#pragma unroll
    for (int j = 0; j < 8; ++j) {
        int c = t + j * 256;
        pr[c] = f2bf(vals[j] * inv);
    }
}

// ---------------------------------------------------------------------------
// Host orchestration
// ---------------------------------------------------------------------------
extern "C" void kernel_launch(void* const* d_in, const int* in_sizes, int n_in,
                              void* d_out, int out_size, void* d_ws, size_t ws_size,
                              hipStream_t stream) {
    (void)in_sizes; (void)n_in; (void)out_size; (void)ws_size;
    const float* x    = (const float*)d_in[0];
    // d_in[1] = attention_mask (causal; applied analytically in softmax)
    const float* wq   = (const float*)d_in[2];
    const float* wk   = (const float*)d_in[3];
    const float* wv   = (const float*)d_in[4];
    const float* wo   = (const float*)d_in[5];
    const float* ln1w = (const float*)d_in[6];
    const float* ln1b = (const float*)d_in[7];
    const float* ln2w = (const float*)d_in[8];
    const float* ln2b = (const float*)d_in[9];
    const float* w1   = (const float*)d_in[10];
    const float* w2   = (const float*)d_in[11];

    constexpr size_t MB = 1024ull * 1024ull;
    char* ws = (char*)d_ws;
    unsigned short* h_bf   = (unsigned short*)(ws + 0);        // 16 MB (also h2)
    float*          q      = (float*)(ws + 16  * MB);          // 48 MB
    float*          k      = (float*)(ws + 64  * MB);          // 48 MB
    float*          v      = (float*)(ws + 112 * MB);          // 32 MB
    unsigned short* ctx_bf = (unsigned short*)(ws + 144 * MB); // 16 MB
    float*          x2     = (float*)(ws + 160 * MB);          // 32 MB
    float*          scores = (float*)(ws + 192 * MB);          // 16 MB
    unsigned short* attn   = (unsigned short*)(ws + 208 * MB); // 8 MB
    unsigned short* qh     = (unsigned short*)(ws + 216 * MB); // 0.75 MB
    unsigned short* kh     = (unsigned short*)(ws + 217 * MB); // 0.75 MB
    unsigned short* vt     = (unsigned short*)(ws + 218 * MB); // 0.5 MB
    unsigned short* g_bf   = (unsigned short*)(ws + 16  * MB); // 64 MB, reuses dead q/k

    const dim3 blk(256);

    // 1) LN1 -> bf16 activations
    ln_kernel<<<SEQ, blk, 0, stream>>>(x, ln1w, ln1b, h_bf);

    // 2) q / k / v projections (A bf16 TDM/async, B fp32 converted, C fp32)
    gemm_wmma_kernel<true, 0, false><<<dim3(QK_DIM / 128, SEQ / 128), blk, 0, stream>>>(
        h_bf, HID, wq, HID, q, QK_DIM, nullptr, SEQ, QK_DIM, HID);
    gemm_wmma_kernel<true, 0, false><<<dim3(QK_DIM / 128, SEQ / 128), blk, 0, stream>>>(
        h_bf, HID, wk, HID, k, QK_DIM, nullptr, SEQ, QK_DIM, HID);
    gemm_wmma_kernel<true, 0, false><<<dim3((NHEAD * D_V) / 128, SEQ / 128), blk, 0, stream>>>(
        h_bf, HID, wv, HID, v, NHEAD * D_V, nullptr, SEQ, NHEAD * D_V, HID);

    // 3) RoPE in place on q/k
    rope_kernel<<<(SEQ * NHEAD * 32) / 256, blk, 0, stream>>>(q, k);

    // 4) attention, one head at a time
    const int gatherBlocks = (2 * SEQ * HEAD_K + D_V * SEQ) / 256; // 4096
    for (int hd = 0; hd < NHEAD; ++hd) {
        gather_head_kernel<<<gatherBlocks, blk, 0, stream>>>(q, k, v, qh, kh, vt, hd);
        // scores = qh @ kh^T (both bf16 -> fully TDM-staged), K = 192
        gemm_wmma_kernel<false, 0, false><<<dim3(SEQ / 128, SEQ / 128), blk, 0, stream>>>(
            qh, HEAD_K, kh, HEAD_K, scores, SEQ, nullptr, SEQ, SEQ, HEAD_K);
        softmax_kernel<<<SEQ, blk, 0, stream>>>(scores, attn);
        // ctx_bf[:, hd*128 : +128] = attn @ v_head  (bf16 out, feeds w_o GEMM as A)
        gemm_wmma_kernel<false, 0, true><<<dim3(1, SEQ / 128), blk, 0, stream>>>(
            attn, SEQ, vt, SEQ, ctx_bf + hd * D_V, NHEAD * D_V, nullptr, SEQ, D_V, SEQ);
    }

    // 5) output projection + residual: x2 = ctx @ wo^T + x
    gemm_wmma_kernel<true, 1, false><<<dim3(HID / 128, SEQ / 128), blk, 0, stream>>>(
        ctx_bf, NHEAD * D_V, wo, NHEAD * D_V, x2, HID, x, SEQ, HID, NHEAD * D_V);

    // 6) LN2 -> bf16 (reuse h_bf)
    ln_kernel<<<SEQ, blk, 0, stream>>>(x2, ln2w, ln2b, h_bf);

    // 7) FFN up + SiLU, bf16 out: g = silu(h2 @ w1^T)
    gemm_wmma_kernel<true, 2, true><<<dim3(FFN_DIM / 128, SEQ / 128), blk, 0, stream>>>(
        h_bf, HID, w1, HID, g_bf, FFN_DIM, nullptr, SEQ, FFN_DIM, HID);

    // 8) FFN down + residual: out = g @ w2^T + x2
    gemm_wmma_kernel<true, 1, false><<<dim3(HID / 128, SEQ / 128), blk, 0, stream>>>(
        g_bf, FFN_DIM, w2, FFN_DIM, (float*)d_out, HID, x2, SEQ, HID, FFN_DIM);
}
